// LinearAttention_39788577030432
// MI455X (gfx1250) — matrix operand order repaired and checked by reading, not compiled, not run on hardware
//
#include <hip/hip_runtime.h>

// ---------------------------------------------------------------------------
// Problem constants (match reference)
// ---------------------------------------------------------------------------
#define B_      4
#define DIM_    256
#define N_      2048
#define HEADS_  8
#define DH_     64
#define HID_    512      // HEADS_*DH_
#define THREEH_ 1536     // 3*HID_
#define SCALE_  0.125f   // DH_^-0.5
#define NKB_    (N_/64)  // key blocks in attention

// ---------------------------------------------------------------------------
// CDNA5 WMMA types (wave32): D(f32 16x16) = A(bf16 16x32) x B(bf16 32x16) + C
// ---------------------------------------------------------------------------
typedef __bf16 v16bf __attribute__((ext_vector_type(16)));
typedef __bf16 v8bf  __attribute__((ext_vector_type(8)));
typedef float  v8f   __attribute__((ext_vector_type(8)));
typedef int    v4i   __attribute__((ext_vector_type(4)));

static __device__ __forceinline__ v8f wmma_bf16(v16bf a, v16bf b, v8f c) {
    // (neg_a, A, neg_b, B, c_mod, C, reuse_a, reuse_b)
    return __builtin_amdgcn_wmma_f32_16x16x32_bf16(false, a, false, b,
                                                   (short)0, c, false, false);
}

// CDNA5 async global->LDS DMA path (ASYNCcnt-tracked), gated on builtin
// availability so the file compiles on any toolchain / host pass.
#if defined(__HIP_DEVICE_COMPILE__) && defined(__gfx1250__) && defined(__has_builtin)
#  if __has_builtin(__builtin_amdgcn_global_load_async_to_lds_b128) && \
      __has_builtin(__builtin_amdgcn_s_wait_asynccnt)
#    define USE_ASYNC_LDS 1
#  endif
#endif
#ifndef USE_ASYNC_LDS
#  define USE_ASYNC_LDS 0
#endif

#if USE_ASYNC_LDS
// param 0 is an AS(1) int4* (clang prints it as "__device__ int4*"),
// param 1 the AS(3) LDS pointer, then imm offset + imm cpol.
typedef __attribute__((address_space(1))) v4i* gv4i_p;
typedef __attribute__((address_space(3))) v4i* sv4i_p;
static __device__ __forceinline__ void async_cp16(const void* g, void* l) {
    __builtin_amdgcn_global_load_async_to_lds_b128((gv4i_p)g, (sv4i_p)l, 0, 0);
}
#endif

// Fragment index helpers (wave32; C/D per ISA 7.12.2, A/B in this file's contiguous k order):
//  A (16x32, MxK): m = lane%16 ; k = (lane/16)*16 + e
//    -> per lane: one contiguous 16-element run at k0 = (lane/16)*16, read as two 8-element halves
//  B (32x16, KxN): n = lane%16 ; k = (lane/16)*16 + e
//    -> per lane: one contiguous 16-element run
//  C/D (16x16):    n = lane%16 ; m = r + (lane/16)*8   (r = vgpr index 0..7)

// ---------------------------------------------------------------------------
// Kernel 1: fp32 -> bf16 weight conversion into workspace
// ---------------------------------------------------------------------------
__global__ void cvt_weights_kernel(const float* __restrict__ wqkv,
                                   const float* __restrict__ wout,
                                   __bf16* __restrict__ wqkv_bf,
                                   __bf16* __restrict__ wout_bf) {
    int i = blockIdx.x * blockDim.x + threadIdx.x;
    if (i < THREEH_ * DIM_) wqkv_bf[i] = (__bf16)wqkv[i];
    if (i < DIM_ * HID_)    wout_bf[i] = (__bf16)wout[i];
}

// ---------------------------------------------------------------------------
// Kernel 2: QKV projection  qkv[b,o,n] = sum_c wqkv[o,c] * x[b,c,n]
// Block: (ntile, otile, b); 128 threads = 4 waves; wave w -> 16 o-rows.
// Q,K stored [b][h][n][c] (Q pre-scaled); V stored c-major [b][h][c][n] so the
// attention kernel can stage it with straight (async) B128 row copies.
// ---------------------------------------------------------------------------
__global__ __launch_bounds__(128)
void qkv_proj_kernel(const float* __restrict__ x,
                     const __bf16* __restrict__ wq,
                     __bf16* __restrict__ Qt,
                     __bf16* __restrict__ Kt,
                     __bf16* __restrict__ Vc) {
    __shared__ __bf16 xt[64][DIM_];       // x tile transposed: [n][c], 32 KB

    const int tid  = threadIdx.x;
    const int lane = tid & 31, wave = tid >> 5;
    const int khalf = lane >> 4, l16 = lane & 15;
    const int ntile = blockIdx.x, otile = blockIdx.y, b = blockIdx.z;

    // Cooperative load of x[b, 0:256, ntile*64 : +64], transposed into LDS
    for (int i = tid; i < DIM_ * 16; i += 128) {      // 256 rows x 16 float4
        int c = i >> 4, n0 = (i & 15) << 2;
        const float* src = x + ((size_t)b * DIM_ + c) * N_ + (size_t)ntile * 64 + n0;
        float4 f = *(const float4*)src;
        float tf[4] = {f.x, f.y, f.z, f.w};
        for (int e = 0; e < 4; ++e) xt[n0 + e][c] = (__bf16)tf[e];
    }
    __syncthreads();

    // A-fragments: 16 weight rows for this wave, full K=256 (8 k-steps of 32)
    const int orow = otile * 64 + wave * 16 + l16;
    const __bf16* wrow = wq + (size_t)orow * DIM_;
    v16bf A[8];
    for (int s = 0; s < 8; ++s)
        for (int hh = 0; hh < 2; ++hh) {
            v8bf v = *(const v8bf*)(wrow + s * 32 + khalf * 16 + hh * 8);
            for (int e = 0; e < 8; ++e) A[s][hh * 8 + e] = v[e];
        }

    const int obase = otile * 64 + wave * 16;
    const int which = obase >> 9;                 // 0=q 1=k 2=v
    const int h     = (obase >> 6) & 7;
    const float scale = (which == 0) ? SCALE_ : 1.0f;
    __bf16* dst = (which == 0) ? Qt : Kt;

    for (int t = 0; t < 4; ++t) {                 // 4 n-subtiles of 16
        v8f acc;
        for (int r = 0; r < 8; ++r) acc[r] = 0.f;
        for (int s = 0; s < 8; ++s) {
            v16bf Bf = *(const v16bf*)(&xt[t * 16 + l16][s * 32 + khalf * 16]);
            acc = wmma_bf16(A[s], Bf, acc);
        }
        const int n  = ntile * 64 + t * 16 + l16;
        const int c0 = wave * 16 + khalf * 8;     // D row r -> c = c0 + r
        if (which == 2) {
            // V c-major: Vc[((b*H + h)*64 + c)*N + n]
            __bf16* vb = Vc + ((size_t)b * HEADS_ + h) * DH_ * N_;
            for (int r = 0; r < 8; ++r)
                vb[(size_t)(c0 + r) * N_ + n] = (__bf16)acc[r];
        } else {
            v8bf o;
            for (int r = 0; r < 8; ++r) o[r] = (__bf16)(acc[r] * scale);
            *(v8bf*)(dst + (((size_t)b * HEADS_ + h) * N_ + n) * DH_ + c0) = o;
        }
    }
}

// ---------------------------------------------------------------------------
// Kernel 3: flash attention per (b, h, 64-query tile); 4 waves x 16 q-rows.
// Double-buffered K/V tiles; async global->LDS DMA when available.
// ---------------------------------------------------------------------------
__global__ __launch_bounds__(128)
void attn_kernel(const __bf16* __restrict__ Qt,
                 const __bf16* __restrict__ Kt,
                 const __bf16* __restrict__ Vc,
                 __bf16* __restrict__ Ot) {
    __shared__ __bf16 k_lds[2][64][DH_];   // [buf][key][c]  16 KB
    __shared__ __bf16 v_lds[2][DH_][64];   // [buf][c][key]  16 KB
    __shared__ __bf16 p_lds[4][16][64];    // per-wave P tile [m][key] 8 KB

    const int tid  = threadIdx.x;
    const int lane = tid & 31, wave = tid >> 5;
    const int khalf = lane >> 4, l16 = lane & 15;
    const int qtile = blockIdx.x, h = blockIdx.y, b = blockIdx.z;

    const size_t bh = (size_t)b * HEADS_ + h;
    const __bf16* Qbh = Qt + bh * N_ * DH_;
    const __bf16* Kbh = Kt + bh * N_ * DH_;
    const __bf16* Vbh = Vc + bh * DH_ * N_;   // c-major

    // Q A-fragments held in registers for entire kernel (K-dim = dh = 2x32)
    const int qrow = qtile * 64 + wave * 16 + l16;
    v16bf qa[2];
    for (int s = 0; s < 2; ++s)
        for (int hh = 0; hh < 2; ++hh) {
            v8bf v = *(const v8bf*)(Qbh + (size_t)qrow * DH_ + s * 32 + khalf * 16 + hh * 8);
            for (int e = 0; e < 8; ++e) qa[s][hh * 8 + e] = v[e];
        }

    v8f Oacc[4];
    for (int t = 0; t < 4; ++t)
        for (int r = 0; r < 8; ++r) Oacc[t][r] = 0.f;
    float mrow[8], lrow[8];
    for (int r = 0; r < 8; ++r) { mrow[r] = -3.0e38f; lrow[r] = 0.f; }

    // ---- tile staging: K tile is a flat 8KB copy; V tile is 64 rows of 128B
    // (stride N_ in global). 512 x 16B transfers each => 4+4 wave-instructions
    // per wave in the async path.
#define STAGE_TILES(KB, BUF)                                                   \
    do {                                                                       \
        const __bf16* ksrc = Kbh + (size_t)(KB) * 64 * DH_;                    \
        const __bf16* vsrc = Vbh + (size_t)(KB) * 64;                          \
        STAGE_BODY(BUF)                                                        \
    } while (0)

#if USE_ASYNC_LDS
#define STAGE_BODY(BUF)                                                        \
        for (int i = tid; i < 512; i += 128)                                   \
            async_cp16(ksrc + (size_t)i * 8, &k_lds[BUF][0][0] + (size_t)i * 8); \
        for (int i = tid; i < 512; i += 128) {                                 \
            int c = i >> 3, ch = (i & 7) << 3;                                 \
            async_cp16(vsrc + (size_t)c * N_ + ch, &v_lds[BUF][c][ch]);        \
        }
#else
#define STAGE_BODY(BUF)                                                        \
        for (int i = tid; i < 512; i += 128)                                   \
            ((uint4*)&k_lds[BUF][0][0])[i] = ((const uint4*)ksrc)[i];          \
        for (int i = tid; i < 512; i += 128) {                                 \
            int c = i >> 3, ch = (i & 7) << 3;                                 \
            *(uint4*)&v_lds[BUF][c][ch] =                                      \
                *(const uint4*)(vsrc + (size_t)c * N_ + ch);                   \
        }
#endif

    STAGE_TILES(0, 0);

    for (int kb = 0; kb < NKB_; ++kb) {
        const int buf = kb & 1;
        if (kb + 1 < NKB_) {
            STAGE_TILES(kb + 1, buf ^ 1);     // overlap next tile with compute
#if !USE_ASYNC_LDS
            __builtin_prefetch(Kbh + (size_t)(kb + 1) * 64 * DH_ + tid * 64, 0, 0);
#endif
        }
#if USE_ASYNC_LDS
        // async loads complete in order: <=8 outstanding => tile kb finished
        if (kb + 1 < NKB_) __builtin_amdgcn_s_wait_asynccnt(8);
        else               __builtin_amdgcn_s_wait_asynccnt(0);
#endif
        __syncthreads();                      // tile kb visible to all waves

        // ---- S = Q K^T : 16q x 64key per wave (8 WMMAs) ----
        v8f S[4];
        for (int kk = 0; kk < 4; ++kk) {
            for (int r = 0; r < 8; ++r) S[kk][r] = 0.f;
            for (int s = 0; s < 2; ++s) {
                v16bf kbf = *(const v16bf*)(&k_lds[buf][kk * 16 + l16][s * 32 + khalf * 16]);
                S[kk] = wmma_bf16(qa[s], kbf, S[kk]);
            }
        }

        // ---- online softmax (row = m = r + khalf*8; cols across 16 lanes) ----
        for (int r = 0; r < 8; ++r) {
            float rm = fmaxf(fmaxf(S[0][r], S[1][r]), fmaxf(S[2][r], S[3][r]));
            for (int off = 1; off < 16; off <<= 1) rm = fmaxf(rm, __shfl_xor(rm, off));
            float mn    = fmaxf(mrow[r], rm);
            float alpha = __expf(mrow[r] - mn);
            mrow[r] = mn;
            float rs = 0.f;
            for (int kk = 0; kk < 4; ++kk) {
                float p = __expf(S[kk][r] - mn);
                S[kk][r] = p;
                rs += p;
            }
            for (int off = 1; off < 16; off <<= 1) rs += __shfl_xor(rs, off);
            lrow[r] = lrow[r] * alpha + rs;
            for (int t = 0; t < 4; ++t) Oacc[t][r] *= alpha;
        }

        // ---- stage P through wave-private LDS: D-layout -> A-layout ----
        for (int kk = 0; kk < 4; ++kk)
            for (int r = 0; r < 8; ++r)
                p_lds[wave][r + khalf * 8][kk * 16 + l16] = (__bf16)S[kk][r];
        asm volatile("" ::: "memory");   // DS ops are in-order within a wave

        v16bf pa[2];
        for (int s = 0; s < 2; ++s)
            for (int hh = 0; hh < 2; ++hh) {
                v8bf v = *(const v8bf*)(&p_lds[wave][l16][s * 32 + khalf * 16 + hh * 8]);
                for (int e = 0; e < 8; ++e) pa[s][hh * 8 + e] = v[e];
            }

        // ---- O += P * V^T : 16q x 64c per wave (8 WMMAs) ----
        for (int ct = 0; ct < 4; ++ct)
            for (int s = 0; s < 2; ++s) {
                v16bf vb = *(const v16bf*)(&v_lds[buf][ct * 16 + l16][s * 32 + khalf * 16]);
                Oacc[ct] = wmma_bf16(pa[s], vb, Oacc[ct]);
            }
        __syncthreads();   // all waves done with buf before it is re-staged
    }

    // ---- epilogue: normalize, store Ot[b][n][h*64+c] bf16 ----
    float invl[8];
    for (int r = 0; r < 8; ++r) invl[r] = 1.f / lrow[r];
    __bf16* Obh = Ot + (size_t)b * N_ * HID_ + h * DH_;
    for (int ct = 0; ct < 4; ++ct)
        for (int r = 0; r < 8; ++r) {
            int q = qtile * 64 + wave * 16 + r + khalf * 8;
            int c = ct * 16 + l16;
            Obh[(size_t)q * HID_ + c] = (__bf16)(Oacc[ct][r] * invl[r]);
        }
#undef STAGE_TILES
#undef STAGE_BODY
}

// ---------------------------------------------------------------------------
// Kernel 4: out[b,o,n] = sum_k wout[o,k] * Ot[b,n,k] + bias[o]
// Block: (ntile, otile, b); wave -> 16 n-rows x 64 o-cols; K = 512.
// ---------------------------------------------------------------------------
__global__ __launch_bounds__(128)
void out_proj_kernel(const __bf16* __restrict__ Ot,
                     const __bf16* __restrict__ wo,
                     const float* __restrict__ bias,
                     float* __restrict__ out) {
    const int tid  = threadIdx.x;
    const int lane = tid & 31, wave = tid >> 5;
    const int khalf = lane >> 4, l16 = lane & 15;
    const int ntile = blockIdx.x, otile = blockIdx.y, b = blockIdx.z;

    const int nrow = ntile * 64 + wave * 16 + l16;
    const __bf16* arow = Ot + ((size_t)b * N_ + nrow) * HID_;

    v8f acc[4];
    for (int t = 0; t < 4; ++t)
        for (int r = 0; r < 8; ++r) acc[t][r] = 0.f;

    for (int s = 0; s < 16; ++s) {                   // 512 / 32 k-steps
        v16bf A;
        for (int hh = 0; hh < 2; ++hh) {
            v8bf v = *(const v8bf*)(arow + s * 32 + khalf * 16 + hh * 8);
            for (int e = 0; e < 8; ++e) A[hh * 8 + e] = v[e];
        }
        for (int t = 0; t < 4; ++t) {
            int o = otile * 64 + t * 16 + l16;
            v16bf Bf = *(const v16bf*)(wo + (size_t)o * HID_ + s * 32 + khalf * 16);
            acc[t] = wmma_bf16(A, Bf, acc[t]);
        }
    }

    for (int t = 0; t < 4; ++t) {
        int o  = otile * 64 + t * 16 + l16;
        float bv = bias[o];
        int n0 = ntile * 64 + wave * 16 + khalf * 8;  // 8 consecutive n per lane
        v8f res;
        for (int r = 0; r < 8; ++r) res[r] = acc[t][r] + bv;
        *(v8f*)(out + ((size_t)b * DIM_ + o) * N_ + n0) = res;
    }
}

// ---------------------------------------------------------------------------
// Host launcher
// ---------------------------------------------------------------------------
extern "C" void kernel_launch(void* const* d_in, const int* in_sizes, int n_in,
                              void* d_out, int out_size, void* d_ws, size_t ws_size,
                              hipStream_t stream) {
    const float* x    = (const float*)d_in[0];   // [B, DIM, N]
    const float* wqkv = (const float*)d_in[1];   // [1536, 256]
    const float* wout = (const float*)d_in[2];   // [256, 512]
    const float* bout = (const float*)d_in[3];   // [256]
    float* out = (float*)d_out;                  // [B, DIM, N]

    char* ws = (char*)d_ws;
    size_t off = 0;
    __bf16* wqkv_bf = (__bf16*)(ws + off); off += (size_t)THREEH_ * DIM_ * 2;  // 768 KB
    __bf16* wout_bf = (__bf16*)(ws + off); off += (size_t)DIM_ * HID_ * 2;     // 256 KB
    __bf16* Qt = (__bf16*)(ws + off); off += (size_t)B_ * HEADS_ * N_ * DH_ * 2; // 8 MB
    __bf16* Kt = (__bf16*)(ws + off); off += (size_t)B_ * HEADS_ * N_ * DH_ * 2; // 8 MB
    __bf16* Vc = (__bf16*)(ws + off); off += (size_t)B_ * HEADS_ * DH_ * N_ * 2; // 8 MB
    __bf16* Ot = (__bf16*)(ws + off); off += (size_t)B_ * N_ * HID_ * 2;         // 8 MB

    cvt_weights_kernel<<<dim3((THREEH_ * DIM_ + 255) / 256), 256, 0, stream>>>(
        wqkv, wout, wqkv_bf, wout_bf);
    qkv_proj_kernel<<<dim3(N_ / 64, THREEH_ / 64, B_), 128, 0, stream>>>(
        x, wqkv_bf, Qt, Kt, Vc);
    attn_kernel<<<dim3(N_ / 64, HEADS_, B_), 128, 0, stream>>>(Qt, Kt, Vc, Ot);
    out_proj_kernel<<<dim3(N_ / 64, DIM_ / 64, B_), 128, 0, stream>>>(
        Ot, wout_bf, bout, out);
}
